// ContrastiveLoss_71124658422130
// MI455X (gfx1250) — compile-verified, matrix operand order checked by reading, not run on hardware
//
#include <hip/hip_runtime.h>
#include <hip/hip_bf16.h>
#include <math.h>

typedef __attribute__((ext_vector_type(16))) _Float16 v16h;
typedef __attribute__((ext_vector_type(8)))  _Float16 v8h;
typedef __attribute__((ext_vector_type(8)))  float    v8f;

#define BATCH 4096
#define DIM   512
#define TWO_N 8192
#define INV_T 10.0f   // 1 / TEMP, TEMP = 0.1
#define TK    32      // K-slice per LDS tile

// ---------------------------------------------------------------------------
// Kernel 0: zero the row-sum accumulator (atomically accumulated later).
// ---------------------------------------------------------------------------
__global__ void zero_kernel(float* __restrict__ rowsum) {
    int i = blockIdx.x * blockDim.x + threadIdx.x;
    if (i < TWO_N) rowsum[i] = 0.0f;
}

// ---------------------------------------------------------------------------
// Kernel 1: L2-normalize emb_i / emb_j rows -> f16 reps[8192][512], and
// compute positives pos[r] = <e_i, e_j> / (|e_i| |e_j|) in fp32.
// One 256-thread block per row; each thread handles 2 of the 512 elements.
// ---------------------------------------------------------------------------
__global__ __launch_bounds__(256) void nrm_kernel(const float* __restrict__ ei,
                                                  const float* __restrict__ ej,
                                                  _Float16* __restrict__ reps,
                                                  float* __restrict__ pos) {
    const int row = blockIdx.x;
    const int t   = threadIdx.x;
    const float2 xi = ((const float2*)(ei + (size_t)row * DIM))[t];
    const float2 xj = ((const float2*)(ej + (size_t)row * DIM))[t];
    float si = xi.x * xi.x + xi.y * xi.y;
    float sj = xj.x * xj.x + xj.y * xj.y;
    float dd = xi.x * xj.x + xi.y * xj.y;

    __shared__ float r0[256], r1[256], r2[256];
    r0[t] = si; r1[t] = sj; r2[t] = dd;
    __syncthreads();
    for (int s = 128; s > 0; s >>= 1) {
        if (t < s) { r0[t] += r0[t + s]; r1[t] += r1[t + s]; r2[t] += r2[t + s]; }
        __syncthreads();
    }
    const float rsi = 1.0f / fmaxf(sqrtf(r0[0]), 1e-12f);
    const float rsj = 1.0f / fmaxf(sqrtf(r1[0]), 1e-12f);
    if (t == 0) pos[row] = r2[0] * rsi * rsj;

    _Float16* zi = reps + (size_t)row * DIM;
    _Float16* zj = reps + (size_t)(row + BATCH) * DIM;
    zi[2 * t]     = (_Float16)(xi.x * rsi);
    zi[2 * t + 1] = (_Float16)(xi.y * rsi);
    zj[2 * t]     = (_Float16)(xj.x * rsj);
    zj[2 * t + 1] = (_Float16)(xj.y * rsj);
}

// ---------------------------------------------------------------------------
// Kernel 2: fused sim = Z Z^T (f16 WMMA, f32 acc) -> exp(sim/T) -> masked
// row-sum.  128x128 tile per 256-thread block (8 waves, 4x2), each wave owns
// a 32x64 subtile = 2x4 WMMA accumulators.  K = 512 in 16 steps of TK=32.
//
// Per K-step the block's unique operand footprint (A 128x32 + B 128x32 f16
// = 16 KB) is staged into LDS with the CDNA5 async global->LDS DMA path
// (global_load_async_to_lds_b128, ASYNCcnt), double-buffered so the next
// tile's DMA overlaps the current tile's WMMAs.  Fragments are then read
// from LDS (ds_load_b128 pairs) in the native WMMA 16-bit A-layout
// (lane[0..15]: M=lane, K=0..7 & 16..23; lane[16..31]: K=8..15 & 24..31);
// since B = Z_cols^T from the same row-major Z, A and B fragment reads are
// identical in shape.
// ---------------------------------------------------------------------------
__global__ __launch_bounds__(256) void simexp_kernel(const _Float16* __restrict__ reps,
                                                     float* __restrict__ rowsum) {
    __shared__ _Float16 lds[2][2][128 * TK];   // [buffer][A|B][row*TK + k] = 32 KB

    const int t      = threadIdx.x;
    const int lane   = t & 31;
    const int wave   = t >> 5;
    const int laneLo = lane & 15;
    const int laneHi = lane >> 4;
    const int waveM  = wave >> 1;   // 0..3
    const int waveN  = wave & 1;    // 0..1
    const int rowBase = blockIdx.y * 128 + waveM * 32;
    const int colBase = blockIdx.x * 128 + waveN * 64;

    const _Float16* Ag = reps + (size_t)(blockIdx.y * 128) * DIM;  // block A rows
    const _Float16* Bg = reps + (size_t)(blockIdx.x * 128) * DIM;  // block B rows

    // Issue one tile's async copy: 512 x 16B chunks per matrix, 2 per thread.
    // chunk ch: row = ch>>2, 16B sub-chunk = ch&3; LDS is row-major, 64B rows.
    auto issue_tile = [&](int k0, int buf) {
#pragma unroll
        for (int c = 0; c < 2; ++c) {
            const int ch  = t + c * 256;
            const int row = ch >> 2;
            const int sub = ch & 3;
            const _Float16* ga = Ag + (size_t)row * DIM + k0 + sub * 8;
            const _Float16* gb = Bg + (size_t)row * DIM + k0 + sub * 8;
            unsigned la = (unsigned)(size_t)(&lds[buf][0][0] + ch * 8);
            unsigned lb = (unsigned)(size_t)(&lds[buf][1][0] + ch * 8);
            asm volatile("global_load_async_to_lds_b128 %0, %1, off"
                         :: "v"(la), "v"(ga) : "memory");
            asm volatile("global_load_async_to_lds_b128 %0, %1, off"
                         :: "v"(lb), "v"(gb) : "memory");
        }
    };

    v8f acc[2][4] = {};

    issue_tile(0, 0);                       // prologue: tile 0 -> buffer 0

    for (int it = 0; it < DIM / TK; ++it) {
        const int cur = it & 1;
        if (it + 1 < DIM / TK) {
            issue_tile((it + 1) * TK, cur ^ 1);       // overlap next tile's DMA
            asm volatile("s_wait_asynccnt 0x4" ::: "memory");  // tile `it` done
        } else {
            asm volatile("s_wait_asynccnt 0x0" ::: "memory");
        }
        __syncthreads();                    // tile `it` visible to all waves

        const _Float16* Al = &lds[cur][0][0];
        const _Float16* Bl = &lds[cur][1][0];
        auto frag = [&](const _Float16* buf, int lrow) -> v16h {
            const _Float16* p = buf + lrow * TK + laneHi * 8;
            v8h lo = *(const v8h*)(p);
            v8h hi = *(const v8h*)(p + 16);
            return __builtin_shufflevector(lo, hi, 0, 1, 2, 3, 4, 5, 6, 7,
                                                   8, 9, 10, 11, 12, 13, 14, 15);
        };

        v16h a[2], b[4];
#pragma unroll
        for (int mf = 0; mf < 2; ++mf)
            a[mf] = frag(Al, waveM * 32 + mf * 16 + laneLo);
#pragma unroll
        for (int nf = 0; nf < 4; ++nf)
            b[nf] = frag(Bl, waveN * 64 + nf * 16 + laneLo);

#pragma unroll
        for (int mf = 0; mf < 2; ++mf)
#pragma unroll
            for (int nf = 0; nf < 4; ++nf)
                acc[mf][nf] = __builtin_amdgcn_wmma_f32_16x16x32_f16(
                    /*neg_a=*/false, a[mf], /*neg_b=*/false, b[nf],
                    /*c_mod=*/(short)0, acc[mf][nf],
                    /*reuse_a=*/false, /*reuse_b=*/false);

        __syncthreads();                    // buffer `cur` free for reuse
    }

    // C/D layout: VGPR r holds (M = r | r+8 per half-wave, N = lane mod 16).
    // exp(sim * 10), zero the main diagonal, sum over N (16 lanes share a row).
#pragma unroll
    for (int mf = 0; mf < 2; ++mf) {
#pragma unroll
        for (int r = 0; r < 8; ++r) {
            const int row = rowBase + mf * 16 + laneHi * 8 + r;
            float s = 0.0f;
#pragma unroll
            for (int nf = 0; nf < 4; ++nf) {
                const int col = colBase + nf * 16 + laneLo;
                const float e = __expf(acc[mf][nf][r] * INV_T);
                s += (row == col) ? 0.0f : e;
            }
            s += __shfl_xor(s, 1, 16);
            s += __shfl_xor(s, 2, 16);
            s += __shfl_xor(s, 4, 16);
            s += __shfl_xor(s, 8, 16);
            if (laneLo == 0) atomicAdd(&rowsum[row], s);
        }
    }
}

// ---------------------------------------------------------------------------
// Kernel 3: loss = mean_i( log(denom_i) - pos[i mod N] / T )
// ---------------------------------------------------------------------------
__global__ __launch_bounds__(256) void loss_kernel(const float* __restrict__ rowsum,
                                                   const float* __restrict__ pos,
                                                   float* __restrict__ out) {
    const int t = threadIdx.x;
    float acc = 0.0f;
    for (int i = t; i < TWO_N; i += 256)
        acc += __logf(rowsum[i]) - INV_T * pos[i & (BATCH - 1)];
    __shared__ float red[256];
    red[t] = acc;
    __syncthreads();
    for (int s = 128; s > 0; s >>= 1) {
        if (t < s) red[t] += red[t + s];
        __syncthreads();
    }
    if (t == 0) out[0] = red[0] / (float)TWO_N;
}

// ---------------------------------------------------------------------------
extern "C" void kernel_launch(void* const* d_in, const int* in_sizes, int n_in,
                              void* d_out, int out_size, void* d_ws, size_t ws_size,
                              hipStream_t stream) {
    const float* ei = (const float*)d_in[0];
    const float* ej = (const float*)d_in[1];
    float* out = (float*)d_out;

    // Workspace: reps f16 [8192][512] (8 MB) | rowsum f32 [8192] | pos f32 [4096]
    _Float16* reps  = (_Float16*)d_ws;
    float* rowsum   = (float*)((char*)d_ws + (size_t)TWO_N * DIM * sizeof(_Float16));
    float* pos      = rowsum + TWO_N;

    zero_kernel<<<TWO_N / 256, 256, 0, stream>>>(rowsum);
    nrm_kernel<<<BATCH, 256, 0, stream>>>(ei, ej, reps, pos);
    dim3 grid(TWO_N / 128, TWO_N / 128);  // 64 x 64 tiles of 128x128
    simexp_kernel<<<grid, 256, 0, stream>>>(reps, rowsum);
    loss_kernel<<<1, 256, 0, stream>>>(rowsum, pos, out);
}
